// NSE_6975026889407
// MI455X (gfx1250) — matrix operand order repaired and checked by reading, not compiled, not run on hardware
//
#include <hip/hip_runtime.h>

#define B_   32
#define L_   1024
#define K_   256
#define K2_  512
#define K4_  1024
#define GRID 64
#define BLOCK 256
#define NWAVES ((GRID * BLOCK) / 32)

typedef __bf16 bf16_t;
typedef __attribute__((ext_vector_type(16))) __bf16 v16bf;
typedef __attribute__((ext_vector_type(8)))  __bf16 v8bf;
typedef __attribute__((ext_vector_type(8)))  float  v8f;

__device__ __forceinline__ bf16_t f2bf(float f) {
  unsigned u = __builtin_bit_cast(unsigned, f);
  unsigned short r = (unsigned short)((u + 0x7FFFu + ((u >> 16) & 1u)) >> 16);
  return __builtin_bit_cast(bf16_t, r);
}

__device__ __forceinline__ float sigm(float v) {
  return 1.0f / (1.0f + __expf(-v));
}

// A fragment: 16x32 bf16 (ISA 7.12.2). Lane row m = lane&15.
// lanes 0-15: K = k0..k0+15; lanes 16-31: K = k0+16..k0+31 (matches B).
// With bf16-staged activations this is one 32B vector load.
__device__ __forceinline__ v16bf frag_a_bf16(const bf16_t* row, int k0, int hsel) {
  return *(const v16bf*)(row + k0 + hsel * 16);
}

// B fragment: 32x16 bf16. Lane col n = lane&15; lanes 0-15 hold K=k0..k0+15,
// lanes 16-31 hold K=k0+16..k0+31 -> one contiguous 32B load from the
// transposed weight row.
__device__ __forceinline__ v16bf frag_b_bf16(const bf16_t* row, int k0, int hsel) {
  return *(const v16bf*)(row + k0 + hsel * 16);
}

__device__ __forceinline__ void gemm_bf16(v8f& acc, const bf16_t* arow,
                                          const bf16_t* brow, int kdim, int hsel) {
  for (int k0 = 0; k0 < kdim; k0 += 32) {
    v16bf a = frag_a_bf16(arow, k0, hsel);
    v16bf b = frag_b_bf16(brow, k0, hsel);
    acc = __builtin_amdgcn_wmma_f32_16x16x32_bf16(false, a, false, b, (short)0, acc,
                                                  false, false);
  }
}

// Device-scope generation barrier. bar[0] = arrive, bar[16] = generation.
__device__ __forceinline__ void grid_sync(unsigned* bar, unsigned target) {
  __threadfence();
  __syncthreads();
  if (threadIdx.x == 0) {
    unsigned* arrive = bar;
    unsigned* gen    = bar + 16;
    unsigned prev = __hip_atomic_fetch_add(arrive, 1u, __ATOMIC_ACQ_REL,
                                           __HIP_MEMORY_SCOPE_AGENT);
    if (prev == (unsigned)(GRID - 1)) {
      __hip_atomic_store(arrive, 0u, __ATOMIC_RELAXED, __HIP_MEMORY_SCOPE_AGENT);
      __hip_atomic_store(gen, target, __ATOMIC_RELEASE, __HIP_MEMORY_SCOPE_AGENT);
    } else {
      while (__hip_atomic_load(gen, __ATOMIC_ACQUIRE, __HIP_MEMORY_SCOPE_AGENT) < target)
        __builtin_amdgcn_s_sleep(1);
    }
  }
  __syncthreads();
  __threadfence();
}

// ---------------------------------------------------------------------------
// Init: mem = x, xbf = bf16(x), zero states, transpose+convert weights, bar=0
// ---------------------------------------------------------------------------
__global__ void nse_init(const float* __restrict__ x,
                         const float* __restrict__ rWx, const float* __restrict__ rWh,
                         const float* __restrict__ wWx, const float* __restrict__ wWh,
                         const float* __restrict__ cW,
                         float* memb, bf16_t* xbf,
                         float* hr, float* cr, float* hw, float* cw,
                         bf16_t* hrbf, bf16_t* hwbf,
                         bf16_t* rWxT, bf16_t* rWhT, bf16_t* wWxT, bf16_t* wWhT,
                         bf16_t* cWT, unsigned* bar) {
  const size_t NMEM = (size_t)B_ * L_ * K_;   // 8388608
  const size_t NST  = (size_t)B_ * K_;        // 8192
  const size_t NR   = (size_t)K_ * K4_;       // 262144 (rWxT/rWhT/wWhT)
  const size_t NW   = (size_t)K2_ * K4_;      // 524288 (wWxT)
  const size_t NC   = (size_t)K2_ * K2_;      // 262144 (cWT)
  size_t stride = (size_t)gridDim.x * blockDim.x;
  size_t t0 = (size_t)blockIdx.x * blockDim.x + threadIdx.x;

  for (size_t i = t0; i < NMEM; i += stride) {
    float v = x[i];
    memb[i] = v;
    xbf[i] = f2bf(v);
  }
  for (size_t i = t0; i < NST; i += stride) {
    hr[i] = 0.f; cr[i] = 0.f; hw[i] = 0.f; cw[i] = 0.f;
    hrbf[i] = f2bf(0.f); hwbf[i] = f2bf(0.f);
  }
  for (size_t i = t0; i < NR; i += stride) {   // row length K_ = 256
    size_t n = i >> 8, k = i & 255u;
    rWxT[i] = f2bf(rWx[k * K4_ + n]);
    rWhT[i] = f2bf(rWh[k * K4_ + n]);
    wWhT[i] = f2bf(wWh[k * K4_ + n]);
  }
  for (size_t i = t0; i < NW; i += stride) {   // row length K2_ = 512
    size_t n = i >> 9, k = i & 511u;
    wWxT[i] = f2bf(wWx[k * K4_ + n]);
  }
  for (size_t i = t0; i < NC; i += stride) {   // row length 512
    size_t n = i >> 9, k = i & 511u;
    cWT[i] = f2bf(cW[k * K2_ + n]);
  }
  if (blockIdx.x == 0 && threadIdx.x < 64) bar[threadIdx.x] = 0u;
}

// ---------------------------------------------------------------------------
// Persistent recurrence kernel
// ---------------------------------------------------------------------------
__global__ void __launch_bounds__(BLOCK) nse_persist(
    const float* __restrict__ rb, const float* __restrict__ wb,
    const float* __restrict__ cb,
    float* memb, const bf16_t* __restrict__ xbf,
    float* gates, float* att,
    float* hr, float* cr, float* hw, float* cw,
    bf16_t* hrbf, bf16_t* hwbf, bf16_t* mrtbf, bf16_t* ctbf,
    const bf16_t* __restrict__ rWxT, const bf16_t* __restrict__ rWhT,
    const bf16_t* __restrict__ wWxT, const bf16_t* __restrict__ wWhT,
    const bf16_t* __restrict__ cWT,
    unsigned* bar, float* __restrict__ out) {
  __shared__ float red[BLOCK];
  const int tid  = threadIdx.x;
  const int gtid = blockIdx.x * BLOCK + tid;
  const int gstride = GRID * BLOCK;
  const int lane = tid & 31;
  const int hsel = lane >> 4;
  const int l15  = lane & 15;
  const int gwave = gtid >> 5;
  unsigned gen = 0;

  for (int t = 0; t < L_; ++t) {
    // ---- P1: reader gates = x_t@rWx + hr@rWh + rb  (32x1024, bf16 WMMA) ----
    for (int tile = gwave; tile < 128; tile += NWAVES) {
      int mT = tile >> 6, nT = tile & 63;
      int mrow = mT * 16 + l15;
      int ncol = nT * 16 + l15;
      v8f acc;
#pragma unroll
      for (int r = 0; r < 8; ++r) acc[r] = 0.0f;
      gemm_bf16(acc, xbf + ((size_t)mrow * L_ + t) * K_, rWxT + (size_t)ncol * K_, K_, hsel);
      gemm_bf16(acc, hrbf + mrow * K_,                   rWhT + (size_t)ncol * K_, K_, hsel);
      float bias = rb[ncol];
#pragma unroll
      for (int r = 0; r < 8; ++r)
        gates[(size_t)(mT * 16 + hsel * 8 + r) * K4_ + ncol] = acc[r] + bias;
    }
    grid_sync(bar, ++gen);

    // ---- P2: reader LSTM elementwise -> hr (f32 + bf16), cr ----
    for (int i = gtid; i < B_ * K_; i += gstride) {
      int b = i >> 8, n = i & 255;
      const float* g = gates + (size_t)b * K4_;
      float ig = g[n], fg = g[K_ + n], gg = g[2 * K_ + n], og = g[3 * K_ + n];
      float cn = sigm(fg) * cr[i] + sigm(ig) * tanhf(gg);
      cr[i] = cn;
      float hn = sigm(og) * tanhf(cn);
      hr[i] = hn;
      hrbf[i] = f2bf(hn);
    }
    grid_sync(bar, ++gen);

    // ---- P3: attention scores s[b,l] = <hr[b], mem[b,l,:]> ----
    for (int i = gtid; i < B_ * L_; i += gstride) {
      int b = i >> 10;
      const float* mrow = memb + (size_t)i * K_;
      const float* h = hr + b * K_;
      float acc = 0.f;
#pragma unroll 4
      for (int k = 0; k < K_; k += 4)
        acc += mrow[k] * h[k] + mrow[k + 1] * h[k + 1] +
               mrow[k + 2] * h[k + 2] + mrow[k + 3] * h[k + 3];
      att[i] = acc;
    }
    grid_sync(bar, ++gen);

    // ---- P4: softmax over L per batch row (one block per row) ----
    if (blockIdx.x < B_) {
      int b = blockIdx.x;
      float v0 = att[b * L_ + tid],       v1 = att[b * L_ + 256 + tid];
      float v2 = att[b * L_ + 512 + tid], v3 = att[b * L_ + 768 + tid];
      float m = fmaxf(fmaxf(v0, v1), fmaxf(v2, v3));
      red[tid] = m; __syncthreads();
      for (int s = 128; s > 0; s >>= 1) {
        if (tid < s) red[tid] = fmaxf(red[tid], red[tid + s]);
        __syncthreads();
      }
      float rmax = red[0]; __syncthreads();
      float e0 = __expf(v0 - rmax), e1 = __expf(v1 - rmax);
      float e2 = __expf(v2 - rmax), e3 = __expf(v3 - rmax);
      red[tid] = e0 + e1 + e2 + e3; __syncthreads();
      for (int s = 128; s > 0; s >>= 1) {
        if (tid < s) red[tid] += red[tid + s];
        __syncthreads();
      }
      float inv = 1.0f / red[0]; __syncthreads();
      att[b * L_ + tid] = e0 * inv;       att[b * L_ + 256 + tid] = e1 * inv;
      att[b * L_ + 512 + tid] = e2 * inv; att[b * L_ + 768 + tid] = e3 * inv;
    }
    grid_sync(bar, ++gen);

    // ---- P5: m_rt[b,k] = sum_l z[b,l]*mem[b,l,k]; staged directly as bf16 ----
    for (int i = gtid; i < B_ * K_; i += gstride) {
      int b = i >> 8, k = i & 255;
      const float* mb = memb + (size_t)b * L_ * K_ + k;
      const float* zb = att + b * L_;
      float acc = 0.f;
      for (int l = 0; l < L_; ++l) acc += zb[l] * mb[(size_t)l * K_];
      mrtbf[i] = f2bf(acc);
    }
    grid_sync(bar, ++gen);

    // ---- P6: composer c_t = [hr, m_rt] @ cW + cb (32x512, WMMA) ----
    for (int tile = gwave; tile < 64; tile += NWAVES) {
      int mT = tile >> 5, nT = tile & 31;
      int mrow = mT * 16 + l15;
      int ncol = nT * 16 + l15;
      v8f acc;
#pragma unroll
      for (int r = 0; r < 8; ++r) acc[r] = 0.0f;
      gemm_bf16(acc, hrbf  + mrow * K_, cWT + (size_t)ncol * K2_,      K_, hsel);
      gemm_bf16(acc, mrtbf + mrow * K_, cWT + (size_t)ncol * K2_ + K_, K_, hsel);
      float bias = cb[ncol];
#pragma unroll
      for (int r = 0; r < 8; ++r)
        ctbf[(size_t)(mT * 16 + hsel * 8 + r) * K2_ + ncol] = f2bf(acc[r] + bias);
    }
    grid_sync(bar, ++gen);

    // ---- P7: writer gates = c_t@wWx + hw@wWh + wb (32x1024, WMMA) ----
    for (int tile = gwave; tile < 128; tile += NWAVES) {
      int mT = tile >> 6, nT = tile & 63;
      int mrow = mT * 16 + l15;
      int ncol = nT * 16 + l15;
      v8f acc;
#pragma unroll
      for (int r = 0; r < 8; ++r) acc[r] = 0.0f;
      gemm_bf16(acc, ctbf + (size_t)mrow * K2_, wWxT + (size_t)ncol * K2_, K2_, hsel);
      gemm_bf16(acc, hwbf + mrow * K_,          wWhT + (size_t)ncol * K_,  K_,  hsel);
      float bias = wb[ncol];
#pragma unroll
      for (int r = 0; r < 8; ++r)
        gates[(size_t)(mT * 16 + hsel * 8 + r) * K4_ + ncol] = acc[r] + bias;
    }
    grid_sync(bar, ++gen);

    // ---- P8: writer LSTM -> hw (f32 + bf16), cw; emit out[:, t, :] ----
    for (int i = gtid; i < B_ * K_; i += gstride) {
      int b = i >> 8, n = i & 255;
      const float* g = gates + (size_t)b * K4_;
      float ig = g[n], fg = g[K_ + n], gg = g[2 * K_ + n], og = g[3 * K_ + n];
      float cn = sigm(fg) * cw[i] + sigm(ig) * tanhf(gg);
      cw[i] = cn;
      float hn = sigm(og) * tanhf(cn);
      hw[i] = hn;
      hwbf[i] = f2bf(hn);
      out[((size_t)b * L_ + t) * K_ + n] = hn;
    }
    grid_sync(bar, ++gen);

    // ---- P9: mem = mem + z*(hw - mem), float4-vectorized ----
    // (No barrier here: the barrier after next-step P1 orders P9 before any
    //  reader of mem, since P1 touches neither mem nor att nor hw.)
    const float4* hw4 = (const float4*)hw;
    float4* mem4 = (float4*)memb;
    for (int i = gtid; i < (B_ * L_ * K_) / 4; i += gstride) {
      int k4 = i & 63;
      int bl = i >> 6;
      int b = bl >> 10;
      float z = att[bl];
      float4 mv = mem4[i];
      float4 hv = hw4[(b << 6) + k4];
      mv.x += z * (hv.x - mv.x);
      mv.y += z * (hv.y - mv.y);
      mv.z += z * (hv.z - mv.z);
      mv.w += z * (hv.w - mv.w);
      mem4[i] = mv;
    }
  }
}

extern "C" void kernel_launch(void* const* d_in, const int* in_sizes, int n_in,
                              void* d_out, int out_size, void* d_ws, size_t ws_size,
                              hipStream_t stream) {
  (void)in_sizes; (void)n_in; (void)out_size; (void)ws_size;
  const float* x   = (const float*)d_in[0];
  const float* rWx = (const float*)d_in[1];
  const float* rWh = (const float*)d_in[2];
  const float* rb  = (const float*)d_in[3];
  const float* wWx = (const float*)d_in[4];
  const float* wWh = (const float*)d_in[5];
  const float* wb  = (const float*)d_in[6];
  const float* cW  = (const float*)d_in[7];
  const float* cb  = (const float*)d_in[8];
  float* out = (float*)d_out;

  char* ws = (char*)d_ws;
  size_t off = 0;
  auto take = [&](size_t bytes) -> void* {
    void* r = ws + off;
    off += (bytes + 255) & ~(size_t)255;
    return r;
  };
  float*   memb  = (float*)take(sizeof(float) * (size_t)B_ * L_ * K_);    // 32 MB
  bf16_t*  xbf   = (bf16_t*)take(sizeof(bf16_t) * (size_t)B_ * L_ * K_);  // 16 MB
  float*   gates = (float*)take(sizeof(float) * (size_t)B_ * K4_);
  float*   att   = (float*)take(sizeof(float) * (size_t)B_ * L_);
  float*   hr    = (float*)take(sizeof(float) * (size_t)B_ * K_);
  float*   cr    = (float*)take(sizeof(float) * (size_t)B_ * K_);
  float*   hw    = (float*)take(sizeof(float) * (size_t)B_ * K_);
  float*   cw    = (float*)take(sizeof(float) * (size_t)B_ * K_);
  bf16_t*  hrbf  = (bf16_t*)take(sizeof(bf16_t) * (size_t)B_ * K_);
  bf16_t*  hwbf  = (bf16_t*)take(sizeof(bf16_t) * (size_t)B_ * K_);
  bf16_t*  mrtbf = (bf16_t*)take(sizeof(bf16_t) * (size_t)B_ * K_);
  bf16_t*  ctbf  = (bf16_t*)take(sizeof(bf16_t) * (size_t)B_ * K2_);
  bf16_t*  rWxT  = (bf16_t*)take(sizeof(bf16_t) * (size_t)K_  * K4_);
  bf16_t*  rWhT  = (bf16_t*)take(sizeof(bf16_t) * (size_t)K_  * K4_);
  bf16_t*  wWxT  = (bf16_t*)take(sizeof(bf16_t) * (size_t)K2_ * K4_);
  bf16_t*  wWhT  = (bf16_t*)take(sizeof(bf16_t) * (size_t)K_  * K4_);
  bf16_t*  cWT   = (bf16_t*)take(sizeof(bf16_t) * (size_t)K2_ * K2_);
  unsigned* bar  = (unsigned*)take(256);

  nse_init<<<dim3(2048), dim3(BLOCK), 0, stream>>>(
      x, rWx, rWh, wWx, wWh, cW, memb, xbf, hr, cr, hw, cw, hrbf, hwbf,
      rWxT, rWhT, wWxT, wWhT, cWT, bar);

  nse_persist<<<dim3(GRID), dim3(BLOCK), 0, stream>>>(
      rb, wb, cb, memb, xbf, gates, att, hr, cr, hw, cw,
      hrbf, hwbf, mrtbf, ctbf,
      rWxT, rWhT, wWxT, wWhT, cWT, bar, out);
}
